// Seq2Seq_15444702396462
// MI455X (gfx1250) — compile-verified, matrix operand order checked by reading, not run on hardware
//
#include <hip/hip_runtime.h>
#include <hip/hip_bf16.h>

// ---------------------------------------------------------------------------
// CDNA5 / gfx1250 seq2seq: bi-LSTM encoder + attention decoder + fused
// vocab log-softmax scoring. bf16 WMMA everywhere; async-to-LDS staging
// (double-buffered) for the dominant vocab GEMM.
// ---------------------------------------------------------------------------

typedef __bf16  v16bf __attribute__((ext_vector_type(16)));
typedef float   v8f   __attribute__((ext_vector_type(8)));

#define E_DIM   256
#define H_DIM   256
#define V_TGT   32000
#define B_SZ    32
#define S_SRC   512
#define S_TGT   128

// ---------------- WMMA helpers (16x16x32 bf16, f32 acc) ---------------------

__device__ __forceinline__ v8f wmma_bf16(v16bf a, v16bf b, v8f c) {
  return __builtin_amdgcn_wmma_f32_16x16x32_bf16(
      /*neg_a=*/false, a, /*neg_b=*/false, b,
      /*c_mod=*/(short)0, c, /*reuse_a=*/false, /*reuse_b=*/false);
}

// 16x32 bf16 fragment from row-major memory (ld in elements).
// lanes 0-15: row m=lane, K={0..7} in v0-3, K={16..23} in v4-7;
// lanes 16-31: row m=lane-16, K={8..15} and K={24..31}.
__device__ __forceinline__ v16bf load_frag(const __bf16* base, int ld) {
  int lane = threadIdx.x & 31;
  int m = lane & 15, h = lane >> 4;
  const __bf16* p = base + (size_t)m * ld;
  v16bf f;
#pragma unroll
  for (int i = 0; i < 8; ++i) {
    f[i]     = p[h * 8 + i];
    f[8 + i] = p[16 + h * 8 + i];
  }
  return f;
}

// 16x16 f32 C tile: VGPR r, lanes 0-15 -> (M=r, N=lane); lanes 16-31 -> (M=8+r).
__device__ __forceinline__ void store_frag(float* p, int ld, v8f c) {
  int lane = threadIdx.x & 31;
  int n = lane & 15, mb = (lane >> 4) * 8;
#pragma unroll
  for (int r = 0; r < 8; ++r) p[(size_t)(mb + r) * ld + n] = c[r];
}

__device__ __forceinline__ float sigf(float x) { return 1.0f / (1.0f + __expf(-x)); }

// gfx1250 async global->LDS copy (ASYNCcnt-tracked). LDS destination is the
// low 32 bits of the generic shared-memory address (ISA 10.2: LDS aperture
// keeps the byte offset in addr[31:0]).
__device__ __forceinline__ void async_copy_b128(unsigned lds_off, const void* gsrc) {
  asm volatile("global_load_async_to_lds_b128 %0, %1, off"
               :: "v"(lds_off), "v"((unsigned long long)gsrc)
               : "memory");
}
__device__ __forceinline__ void wait_async0() {
  asm volatile("s_wait_asynccnt 0x0" ::: "memory");
}

// ---------------- small utility kernels -------------------------------------

__global__ void f2bf_kernel(const float* __restrict__ x, __bf16* __restrict__ y, int n) {
  int i = blockIdx.x * blockDim.x + threadIdx.x;
  if (i < n) y[i] = (__bf16)x[i];
}

__global__ void embed_kernel(const int* __restrict__ tok, const float* __restrict__ emb,
                             __bf16* __restrict__ out, int nrows) {
  int row = blockIdx.x;
  if (row >= nrows) return;
  int t = tok[row];
  for (int j = threadIdx.x; j < E_DIM; j += blockDim.x)
    out[(size_t)row * E_DIM + j] = (__bf16)emb[(size_t)t * E_DIM + j];
}

__global__ void hcat_kernel(const float* __restrict__ hF, const float* __restrict__ hB,
                            __bf16* __restrict__ hcat) {
  int i = blockIdx.x * blockDim.x + threadIdx.x;   // 32*512
  if (i >= B_SZ * 2 * H_DIM) return;
  int b = i >> 9, j = i & 511;
  hcat[i] = (__bf16)((j < H_DIM) ? hF[b * H_DIM + j] : hB[b * H_DIM + j - H_DIM]);
}

__global__ void zero_kernel(float* p, int n) {
  int i = blockIdx.x * blockDim.x + threadIdx.x;
  if (i < n) p[i] = 0.0f;
}

// -------- generic GEMM: C(MxN,f32) = A(MxK,bf16) * W(NxK,bf16)^T ------------
// Each wave computes a 16x64 strip (4 N-tiles), reusing every A fragment 4x.

__global__ __launch_bounds__(256) void gemm_nt_bf16(
    const __bf16* __restrict__ A, const __bf16* __restrict__ W,
    float* __restrict__ C, int M, int N, int K) {
  int wave  = blockIdx.x * (blockDim.x >> 5) + (threadIdx.x >> 5);
  int ngrp  = N >> 6;                      // groups of 4 N-tiles
  int waves = (M >> 4) * ngrp;
  if (wave >= waves) return;               // wave-uniform
  int tm = wave / ngrp, tg = wave % ngrp;
  const __bf16* a0 = A + (size_t)tm * 16 * K;
  v8f acc[4] = {};
  for (int k = 0; k < K; k += 32) {
    v16bf a = load_frag(a0 + k, K);
#pragma unroll
    for (int j = 0; j < 4; ++j)
      acc[j] = wmma_bf16(a, load_frag(W + (size_t)(tg * 4 + j) * 16 * K + k, K), acc[j]);
  }
#pragma unroll
  for (int j = 0; j < 4; ++j)
    store_frag(C + (size_t)tm * 16 * N + (tg * 4 + j) * 16, N, acc[j]);
}

// ---------------- persistent encoder LSTM (one WG per direction) ------------

__global__ __launch_bounds__(1024) void enc_lstm_kernel(
    const __bf16* __restrict__ whh_f, const __bf16* __restrict__ whh_b,
    const float* __restrict__ bias_f, const float* __restrict__ bias_b,
    const float* __restrict__ gx_f,  const float* __restrict__ gx_b,
    __bf16* __restrict__ enc, float* __restrict__ hF, float* __restrict__ hB) {
  extern __shared__ char smem[];
  __bf16* h_bf  = (__bf16*)smem;                       // 32*256 bf16
  float*  gates = (float*)(smem + B_SZ * H_DIM * 2);   // 32*1024 f32

  int dir = blockIdx.x;
  const __bf16* whh  = dir ? whh_b  : whh_f;
  const float*  bias = dir ? bias_b : bias_f;
  const float*  gx   = dir ? gx_b   : gx_f;
  float*        hfin = dir ? hB     : hF;

  int tid = threadIdx.x, wv = tid >> 5;
  float hreg[8], creg[8];
#pragma unroll
  for (int i = 0; i < 8; ++i) { hreg[i] = 0.f; creg[i] = 0.f; }
  for (int i = tid; i < B_SZ * H_DIM; i += 1024) h_bf[i] = (__bf16)0.f;
  __syncthreads();

  for (int s = 0; s < S_SRC; ++s) {
    int t = dir ? (S_SRC - 1 - s) : s;
    // gates = h(32x256) @ whh^T(1024x256): wave -> 16x64 strip
    {
      int tm = wv >> 4, tn0 = (wv & 15) * 4;
      v8f acc[4] = {};
      const __bf16* a0 = h_bf + (size_t)tm * 16 * H_DIM;
      for (int k = 0; k < H_DIM; k += 32) {
        v16bf a = load_frag(a0 + k, H_DIM);
#pragma unroll
        for (int j = 0; j < 4; ++j)
          acc[j] = wmma_bf16(a, load_frag(whh + (size_t)(tn0 + j) * 16 * H_DIM + k, H_DIM), acc[j]);
      }
#pragma unroll
      for (int j = 0; j < 4; ++j)
        store_frag(gates + (size_t)tm * 16 * 1024 + (tn0 + j) * 16, 1024, acc[j]);
    }
    __syncthreads();
    // elementwise LSTM: 8192 (b,j) pairs, 8 per thread
#pragma unroll
    for (int i = 0; i < 8; ++i) {
      int p = tid * 8 + i, b = p >> 8, j = p & 255;
      size_t grow = ((size_t)t * B_SZ + b) * 1024;
      float gi = gates[b * 1024 + j]       + gx[grow + j]       + bias[j];
      float gf = gates[b * 1024 + 256 + j] + gx[grow + 256 + j] + bias[256 + j];
      float gg = gates[b * 1024 + 512 + j] + gx[grow + 512 + j] + bias[512 + j];
      float go = gates[b * 1024 + 768 + j] + gx[grow + 768 + j] + bias[768 + j];
      float cN = sigf(gf) * creg[i] + sigf(gi) * tanhf(gg);
      float hN = sigf(go) * tanhf(cN);
      creg[i] = cN; hreg[i] = hN;
      enc[((size_t)b * S_SRC + t) * (2 * H_DIM) + dir * H_DIM + j] = (__bf16)hN;
    }
#pragma unroll
    for (int i = 0; i < 8; ++i) h_bf[tid * 8 + i] = (__bf16)hreg[i];
    __syncthreads();
  }
#pragma unroll
  for (int i = 0; i < 8; ++i) hfin[tid * 8 + i] = hreg[i];
}

// ---------------- persistent attention decoder (single WG) ------------------

__global__ __launch_bounds__(1024) void dec_kernel(
    const __bf16* __restrict__ wih_d, const __bf16* __restrict__ whh_d,
    const float* __restrict__ b_d,    const __bf16* __restrict__ wcomb,
    const __bf16* __restrict__ Y,     const float* __restrict__ proj,
    const __bf16* __restrict__ enc,   const float* __restrict__ h0,
    const float* __restrict__ c0,     __bf16* __restrict__ hid) {
  extern __shared__ char smem[];
  float*  gates = (float*)smem;               // 32*1024 f32  (128KB)
  float*  e     = gates;                      // alias: 32*512 f32
  float*  o_pre = gates + B_SZ * S_SRC;       // alias: 32*256 f32
  __bf16* i_bf  = (__bf16*)(smem + 131072);                     // 32*512
  __bf16* h_bf  = (__bf16*)(smem + 131072 + 32768);             // 32*256
  float*  h_s   = (float*) (smem + 131072 + 32768 + 16384);     // 32*256
  __bf16* o_bf  = (__bf16*)(smem + 131072 + 32768 + 16384 + 32768);          // 32*256
  __bf16* hctx  = (__bf16*)(smem + 131072 + 32768 + 16384 + 32768 + 16384);  // 32*768

  int tid = threadIdx.x, wv = tid >> 5, lane = tid & 31;
  float creg[8];
#pragma unroll
  for (int i = 0; i < 8; ++i) {
    int p = tid * 8 + i;
    float hv = h0[p];
    h_s[p] = hv; h_bf[p] = (__bf16)hv;
    creg[i] = c0[p];
    o_bf[p] = (__bf16)0.f;
  }
  __syncthreads();

  for (int t = 0; t < S_TGT - 1; ++t) {
    // 1. i_t = [y_t, o_prev]  (32x512)
#pragma unroll
    for (int i = 0; i < 16; ++i) {
      int p = tid * 16 + i, b = p >> 9, j = p & 511;
      i_bf[p] = (j < E_DIM) ? Y[((size_t)t * B_SZ + b) * E_DIM + j]
                            : o_bf[b * H_DIM + (j - E_DIM)];
    }
    __syncthreads();
    // 2. gates = i_t @ Wih_d^T + h @ Whh_d^T : wave -> 16x64 strip
    {
      int tm = wv >> 4, tn0 = (wv & 15) * 4;
      v8f acc[4] = {};
      const __bf16* a0 = i_bf + (size_t)tm * 16 * 512;
      for (int k = 0; k < 512; k += 32) {
        v16bf a = load_frag(a0 + k, 512);
#pragma unroll
        for (int j = 0; j < 4; ++j)
          acc[j] = wmma_bf16(a, load_frag(wih_d + (size_t)(tn0 + j) * 16 * 512 + k, 512), acc[j]);
      }
      const __bf16* a1 = h_bf + (size_t)tm * 16 * H_DIM;
      for (int k = 0; k < H_DIM; k += 32) {
        v16bf a = load_frag(a1 + k, H_DIM);
#pragma unroll
        for (int j = 0; j < 4; ++j)
          acc[j] = wmma_bf16(a, load_frag(whh_d + (size_t)(tn0 + j) * 16 * H_DIM + k, H_DIM), acc[j]);
      }
#pragma unroll
      for (int j = 0; j < 4; ++j)
        store_frag(gates + (size_t)tm * 16 * 1024 + (tn0 + j) * 16, 1024, acc[j]);
    }
    __syncthreads();
    // 3. LSTM cell
    float hnew[8];
#pragma unroll
    for (int i = 0; i < 8; ++i) {
      int p = tid * 8 + i, b = p >> 8, j = p & 255;
      float gi = gates[b * 1024 + j]       + b_d[j];
      float gf = gates[b * 1024 + 256 + j] + b_d[256 + j];
      float gg = gates[b * 1024 + 512 + j] + b_d[512 + j];
      float go = gates[b * 1024 + 768 + j] + b_d[768 + j];
      float cN = sigf(gf) * creg[i] + sigf(gi) * tanhf(gg);
      creg[i] = cN; hnew[i] = sigf(go) * tanhf(cN);
    }
    __syncthreads();          // gates reads done (e aliases gates)
#pragma unroll
    for (int i = 0; i < 8; ++i) {
      int p = tid * 8 + i;
      h_s[p] = hnew[i]; h_bf[p] = (__bf16)hnew[i];
    }
    __syncthreads();
    // 4. e[b,s] = proj[b,s,:] . h[b,:]
#pragma unroll
    for (int i = 0; i < 16; ++i) {
      int p = tid * 16 + i, b = p >> 9, s = p & 511;
      const float* pr = proj + ((size_t)b * S_SRC + s) * H_DIM;
      const float* hh = h_s + b * H_DIM;
      float acc = 0.f;
      for (int k = 0; k < H_DIM; ++k) acc += pr[k] * hh[k];
      e[b * S_SRC + s] = acc;
    }
    __syncthreads();
    // 5. softmax over s (one wave per batch row)
    {
      int b = wv;
      float m = -1e30f;
      for (int s = lane; s < S_SRC; s += 32) m = fmaxf(m, e[b * S_SRC + s]);
#pragma unroll
      for (int d = 16; d > 0; d >>= 1) m = fmaxf(m, __shfl_xor(m, d, 32));
      float sum = 0.f;
      for (int s = lane; s < S_SRC; s += 32) {
        float v = __expf(e[b * S_SRC + s] - m);
        e[b * S_SRC + s] = v; sum += v;
      }
#pragma unroll
      for (int d = 16; d > 0; d >>= 1) sum += __shfl_xor(sum, d, 32);
      float inv = 1.f / sum;
      for (int s = lane; s < S_SRC; s += 32) e[b * S_SRC + s] *= inv;
    }
    __syncthreads();
    // 6. ctx[b,d] = sum_s a[b,s] * enc[b,s,d]; build [h, ctx] (32x768)
#pragma unroll
    for (int i = 0; i < 16; ++i) {
      int p = tid * 16 + i, b = p >> 9, d = p & 511;
      float acc = 0.f;
      for (int s = 0; s < S_SRC; ++s)
        acc += e[b * S_SRC + s] * (float)enc[((size_t)b * S_SRC + s) * 512 + d];
      hctx[b * 768 + 256 + d] = (__bf16)acc;
    }
#pragma unroll
    for (int i = 0; i < 8; ++i) {
      int p = tid * 8 + i, b = p >> 8, j = p & 255;
      hctx[b * 768 + j] = (__bf16)h_s[b * H_DIM + j];
    }
    __syncthreads();
    // 7. o_pre = [h,ctx](32x768) @ Wcomb^T(256x768): 32 tiles, one per wave
    {
      int tm = wv >> 4, tn = wv & 15;
      v8f acc = {};
      const __bf16* a0 = hctx  + (size_t)tm * 16 * 768;
      const __bf16* b0 = wcomb + (size_t)tn * 16 * 768;
      for (int k = 0; k < 768; k += 32)
        acc = wmma_bf16(load_frag(a0 + k, 768), load_frag(b0 + k, 768), acc);
      store_frag(o_pre + (size_t)tm * 16 * H_DIM + tn * 16, H_DIM, acc);
    }
    __syncthreads();
    // 8. o_t = tanh(o_pre)
#pragma unroll
    for (int i = 0; i < 8; ++i) {
      int p = tid * 8 + i, b = p >> 8, j = p & 255;
      float v = tanhf(o_pre[b * H_DIM + j]);
      o_bf[p] = (__bf16)v;
      hid[((size_t)t * B_SZ + b) * H_DIM + j] = (__bf16)v;
    }
    __syncthreads();
  }
}

// -------- fused vocab GEMM + online logsumexp + score -----------------------
// Block = 128 rows (8 waves x 16-row A tiles, A frags in registers).
// All 8 waves share each 16x256 Wvoc tile, double-buffered in LDS via
// global_load_async_to_lds_b128 (ASYNCcnt), overlapping copy with WMMA.

__global__ __launch_bounds__(256) void vocab_kernel(
    const __bf16* __restrict__ hid, const __bf16* __restrict__ wvoc,
    const int* __restrict__ target, float* __restrict__ out, int nrows) {
  __shared__ __bf16 bstage[2][16 * H_DIM];   // 2 x 8KB
  __shared__ float  lp[128];
  __shared__ int    idxs[128];

  int tid = threadIdx.x, wv = tid >> 5, lane = tid & 31;
  int r0 = blockIdx.x * 128;

  if (tid < 128) {
    int row  = r0 + tid;
    int rowc = row < nrows ? row : nrows - 1;
    int t = rowc >> 5, b = rowc & 31;
    idxs[tid] = target[(t + 1) * B_SZ + b];   // target[1:]
    lp[tid]   = 0.f;
  }

  // A fragments for this wave's 16 rows (clamped; invalid rows discarded later)
  int rowbase = r0 + wv * 16;
  if (rowbase > nrows - 16) rowbase = nrows - 16;
  const __bf16* a0 = hid + (size_t)rowbase * H_DIM;
  v16bf afr[8];
#pragma unroll
  for (int k = 0; k < 8; ++k) afr[k] = load_frag(a0 + k * 32, H_DIM);

  unsigned lds0 = (unsigned)(unsigned long long)(void*)&bstage[0][0];

  // stage tile 0 (8KB: 2 x b128 per thread)
  async_copy_b128(lds0 + tid * 16u,          (const char*)wvoc + tid * 16);
  async_copy_b128(lds0 + 4096u + tid * 16u,  (const char*)wvoc + 4096 + tid * 16);
  wait_async0();
  __syncthreads();

  float m[8], s[8];
#pragma unroll
  for (int r = 0; r < 8; ++r) { m[r] = -1e30f; s[r] = 0.f; }
  int mb = (lane >> 4) * 8, n = lane & 15;

  const int NT = V_TGT / 16;                 // 2000 vocab tiles
  for (int nt = 0; nt < NT; ++nt) {
    int p = nt & 1;
    if (nt + 1 < NT) {                       // prefetch next tile into other buffer
      const char* src = (const char*)(wvoc + (size_t)(nt + 1) * 16 * H_DIM);
      unsigned dst = lds0 + (unsigned)((p ^ 1) * 8192);
      async_copy_b128(dst + tid * 16u,         src + tid * 16);
      async_copy_b128(dst + 4096u + tid * 16u, src + 4096 + tid * 16);
    }
    const __bf16* bt = &bstage[p][0];
    v8f acc = {};
#pragma unroll
    for (int k = 0; k < 8; ++k)
      acc = wmma_bf16(afr[k], load_frag(bt + k * 32, H_DIM), acc);
    int v = nt * 16 + n;
#pragma unroll
    for (int r = 0; r < 8; ++r) {
      float x = acc[r];
      if (v == idxs[wv * 16 + mb + r]) lp[wv * 16 + mb + r] = x;
      float nm = fmaxf(m[r], x);
      s[r] = s[r] * __expf(m[r] - nm) + __expf(x - nm);
      m[r] = nm;
    }
    wait_async0();
    __syncthreads();
  }
  // reduce (m,s) across the 16 lanes sharing the same rows (single wave owns rows)
#pragma unroll
  for (int d = 8; d > 0; d >>= 1) {
#pragma unroll
    for (int r = 0; r < 8; ++r) {
      float om = __shfl_xor(m[r], d, 32);
      float os = __shfl_xor(s[r], d, 32);
      float nm = fmaxf(m[r], om);
      s[r] = s[r] * __expf(m[r] - nm) + os * __expf(om - nm);
      m[r] = nm;
    }
  }
  if ((lane & 15) == 0) {
#pragma unroll
    for (int r = 0; r < 8; ++r) {
      int jrow = wv * 16 + mb + r;
      int row  = r0 + jrow;
      if (row < nrows) {
        float lse  = m[r] + __logf(s[r]);
        float mask = (idxs[jrow] != 0) ? 1.f : 0.f;   // PAD == 0
        atomicAdd(&out[row & 31], (lp[jrow] - lse) * mask);
      }
    }
  }
}

// ---------------------------------------------------------------------------

extern "C" void kernel_launch(void* const* d_in, const int* in_sizes, int n_in,
                              void* d_out, int out_size, void* d_ws, size_t ws_size,
                              hipStream_t stream) {
  const int*   src     = (const int*)  d_in[0];
  const int*   tgt     = (const int*)  d_in[1];
  const float* src_emb = (const float*)d_in[2];
  const float* tgt_emb = (const float*)d_in[3];
  const float* Wih_f   = (const float*)d_in[4];
  const float* Whh_f   = (const float*)d_in[5];
  const float* b_f     = (const float*)d_in[6];
  const float* Wih_b   = (const float*)d_in[7];
  const float* Whh_b   = (const float*)d_in[8];
  const float* b_b     = (const float*)d_in[9];
  const float* Wih_d   = (const float*)d_in[10];
  const float* Whh_d   = (const float*)d_in[11];
  const float* b_d     = (const float*)d_in[12];
  const float* Wh      = (const float*)d_in[13];
  const float* Wc      = (const float*)d_in[14];
  const float* Watt    = (const float*)d_in[15];
  const float* Wcomb   = (const float*)d_in[16];
  const float* Wvoc    = (const float*)d_in[17];
  float* out = (float*)d_out;
  (void)in_sizes; (void)n_in; (void)out_size; (void)ws_size;

  size_t off = 0;
  auto alloc = [&](size_t bytes) -> void* {
    off = (off + 255) & ~(size_t)255;
    void* p = (char*)d_ws + off;
    off += bytes;
    return p;
  };

  __bf16* wihf = (__bf16*)alloc((size_t)1024 * 256 * 2);
  __bf16* whhf = (__bf16*)alloc((size_t)1024 * 256 * 2);
  __bf16* wihb = (__bf16*)alloc((size_t)1024 * 256 * 2);
  __bf16* whhb = (__bf16*)alloc((size_t)1024 * 256 * 2);
  __bf16* wihd = (__bf16*)alloc((size_t)1024 * 512 * 2);
  __bf16* whhd = (__bf16*)alloc((size_t)1024 * 256 * 2);
  __bf16* wh   = (__bf16*)alloc((size_t)256 * 512 * 2);
  __bf16* wc   = (__bf16*)alloc((size_t)256 * 512 * 2);
  __bf16* watt = (__bf16*)alloc((size_t)256 * 512 * 2);
  __bf16* wcmb = (__bf16*)alloc((size_t)256 * 768 * 2);
  __bf16* wvoc = (__bf16*)alloc((size_t)V_TGT * 256 * 2);
  __bf16* Xbf  = (__bf16*)alloc((size_t)S_SRC * B_SZ * E_DIM * 2);
  __bf16* Ybf  = (__bf16*)alloc((size_t)(S_TGT - 1) * B_SZ * E_DIM * 2);
  __bf16* hcat = (__bf16*)alloc((size_t)B_SZ * 512 * 2);
  __bf16* enc  = (__bf16*)alloc((size_t)B_SZ * S_SRC * 512 * 2);
  __bf16* hid  = (__bf16*)alloc((size_t)(S_TGT - 1) * B_SZ * H_DIM * 2);
  float*  gxf  = (float*)alloc((size_t)S_SRC * B_SZ * 1024 * 4);
  float*  gxb  = (float*)alloc((size_t)S_SRC * B_SZ * 1024 * 4);
  float*  proj = (float*)alloc((size_t)B_SZ * S_SRC * H_DIM * 4);
  float*  hF   = (float*)alloc((size_t)B_SZ * H_DIM * 4);
  float*  hB   = (float*)alloc((size_t)B_SZ * H_DIM * 4);
  float*  h_t  = (float*)alloc((size_t)B_SZ * H_DIM * 4);
  float*  c_t  = (float*)alloc((size_t)B_SZ * H_DIM * 4);

  auto cvt = [&](const float* s_, __bf16* d_, int n) {
    f2bf_kernel<<<(n + 255) / 256, 256, 0, stream>>>(s_, d_, n);
  };
  cvt(Wih_f, wihf, 1024 * 256);  cvt(Whh_f, whhf, 1024 * 256);
  cvt(Wih_b, wihb, 1024 * 256);  cvt(Whh_b, whhb, 1024 * 256);
  cvt(Wih_d, wihd, 1024 * 512);  cvt(Whh_d, whhd, 1024 * 256);
  cvt(Wh, wh, 256 * 512);        cvt(Wc, wc, 256 * 512);
  cvt(Watt, watt, 256 * 512);    cvt(Wcomb, wcmb, 256 * 768);
  cvt(Wvoc, wvoc, V_TGT * 256);

  embed_kernel<<<S_SRC * B_SZ, 256, 0, stream>>>(src, src_emb, Xbf, S_SRC * B_SZ);
  embed_kernel<<<(S_TGT - 1) * B_SZ, 256, 0, stream>>>(tgt, tgt_emb, Ybf,
                                                       (S_TGT - 1) * B_SZ);

  // gx = X @ Wih^T  (16384 x 1024, K=256), both directions
  {
    int M = S_SRC * B_SZ, N = 1024, K = E_DIM;
    int waves = (M / 16) * (N / 64);
    gemm_nt_bf16<<<(waves + 7) / 8, 256, 0, stream>>>(Xbf, wihf, gxf, M, N, K);
    gemm_nt_bf16<<<(waves + 7) / 8, 256, 0, stream>>>(Xbf, wihb, gxb, M, N, K);
  }

  // persistent bi-LSTM encoder
  {
    size_t lds = (size_t)B_SZ * H_DIM * 2 + (size_t)B_SZ * 1024 * 4;  // 147456
    hipFuncSetAttribute((const void*)enc_lstm_kernel,
                        hipFuncAttributeMaxDynamicSharedMemorySize, (int)lds);
    enc_lstm_kernel<<<2, 1024, lds, stream>>>(whhf, whhb, b_f, b_b, gxf, gxb,
                                              enc, hF, hB);
  }

  hcat_kernel<<<(B_SZ * 512 + 255) / 256, 256, 0, stream>>>(hF, hB, hcat);
  {
    int M = B_SZ, N = H_DIM, K = 512;
    int waves = (M / 16) * (N / 64);
    gemm_nt_bf16<<<(waves + 7) / 8, 256, 0, stream>>>(hcat, wh, h_t, M, N, K);
    gemm_nt_bf16<<<(waves + 7) / 8, 256, 0, stream>>>(hcat, wc, c_t, M, N, K);
  }

  // enc_hiddens_proj = enc @ Watt^T  (16384 x 256, K=512)
  {
    int M = B_SZ * S_SRC, N = H_DIM, K = 512;
    int waves = (M / 16) * (N / 64);
    gemm_nt_bf16<<<(waves + 7) / 8, 256, 0, stream>>>(enc, watt, proj, M, N, K);
  }

  // persistent attention decoder
  {
    size_t lds = 131072 + 32768 + 16384 + 32768 + 16384 + 49152;  // 278528
    hipFuncSetAttribute((const void*)dec_kernel,
                        hipFuncAttributeMaxDynamicSharedMemorySize, (int)lds);
    dec_kernel<<<1, 1024, lds, stream>>>(wihd, whhd, b_d, wcmb, Ybf, proj,
                                         enc, h_t, c_t, hid);
  }

  // fused vocab GEMM + log-softmax + score accumulation
  zero_kernel<<<1, 64, 0, stream>>>(out, B_SZ);
  {
    int nrows  = (S_TGT - 1) * B_SZ;                  // 4064
    int blocks = (nrows + 127) / 128;                 // 32
    vocab_kernel<<<blocks, 256, 0, stream>>>(hid, wvoc, tgt, out, nrows);
  }
}
